// HAILPModel_22574348108497
// MI455X (gfx1250) — matrix operand order repaired and checked by reading, not compiled, not run on hardware
//
#include <hip/hip_runtime.h>
#include <hip/hip_bf16.h>
#include <math.h>

typedef unsigned short u16;
typedef __bf16 v16bf __attribute__((ext_vector_type(16)));
typedef __bf16 v8bf  __attribute__((ext_vector_type(8)));
typedef float  v8f   __attribute__((ext_vector_type(8)));

#define Bn    8
#define Tt    1024
#define BT    8192
#define Cc    512
#define Vv    8000
#define Rr    64
#define ARr   32
#define Ll    12
#define GRP   4
#define FFNc  2048
#define NCH   64   // WKV chunks along T
#define CL    16   // WKV chunk length (NCH*CL == Tt)

__device__ __forceinline__ u16 f2bf(float f) {
  unsigned u = __float_as_uint(f);
  u += 0x7FFFu + ((u >> 16) & 1u);
  return (u16)(u >> 16);
}

// ---------------- embedding gather ----------------
__global__ void gather_embed(const int* __restrict__ ids, const float* __restrict__ embed,
                             float* __restrict__ x) {
  int id = blockIdx.x * blockDim.x + threadIdx.x;
  if (id >= BT * Cc) return;
  int r = id / Cc, c = id % Cc;
  x[id] = embed[(size_t)ids[r] * Cc + c];
}

// ---------------- fused ln1 -> tm_ln (C = 512, 256 threads/row) ----------------
__global__ void ln2x_kernel(const float* __restrict__ x,
                            const float* __restrict__ g1, const float* __restrict__ b1,
                            const float* __restrict__ g2, const float* __restrict__ b2,
                            float* __restrict__ out) {
  __shared__ float ssum[256];
  __shared__ float ssq[256];
  int t = threadIdx.x;
  const float* row = x + (size_t)blockIdx.x * Cc;
  float v0 = row[t], v1 = row[t + 256];
  ssum[t] = v0 + v1;
  ssq[t]  = v0 * v0 + v1 * v1;
  __syncthreads();
  for (int o = 128; o > 0; o >>= 1) {
    if (t < o) { ssum[t] += ssum[t + o]; ssq[t] += ssq[t + o]; }
    __syncthreads();
  }
  float m = ssum[0] * (1.f / Cc);
  float var = ssq[0] * (1.f / Cc) - m * m;
  float rs = rsqrtf(var + 1e-5f);
  float y0 = (v0 - m) * rs * g1[t]       + b1[t];
  float y1 = (v1 - m) * rs * g1[t + 256] + b1[t + 256];
  __syncthreads();
  ssum[t] = y0 + y1;
  ssq[t]  = y0 * y0 + y1 * y1;
  __syncthreads();
  for (int o = 128; o > 0; o >>= 1) {
    if (t < o) { ssum[t] += ssum[t + o]; ssq[t] += ssq[t + o]; }
    __syncthreads();
  }
  m = ssum[0] * (1.f / Cc);
  var = ssq[0] * (1.f / Cc) - m * m;
  rs = rsqrtf(var + 1e-5f);
  float* orow = out + (size_t)blockIdx.x * Cc;
  orow[t]       = (y0 - m) * rs * g2[t]       + b2[t];
  orow[t + 256] = (y1 - m) * rs * g2[t + 256] + b2[t + 256];
}

// ---------------- single layernorm, optional f32 / bf16 outputs ----------------
__global__ void ln_kernel(const float* __restrict__ x,
                          const float* __restrict__ g, const float* __restrict__ b,
                          float* __restrict__ outF, u16* __restrict__ outB) {
  __shared__ float ssum[256];
  __shared__ float ssq[256];
  int t = threadIdx.x;
  const float* row = x + (size_t)blockIdx.x * Cc;
  float v0 = row[t], v1 = row[t + 256];
  ssum[t] = v0 + v1;
  ssq[t]  = v0 * v0 + v1 * v1;
  __syncthreads();
  for (int o = 128; o > 0; o >>= 1) {
    if (t < o) { ssum[t] += ssum[t + o]; ssq[t] += ssq[t + o]; }
    __syncthreads();
  }
  float m = ssum[0] * (1.f / Cc);
  float var = ssq[0] * (1.f / Cc) - m * m;
  float rs = rsqrtf(var + 1e-5f);
  float y0 = (v0 - m) * rs * g[t]       + b[t];
  float y1 = (v1 - m) * rs * g[t + 256] + b[t + 256];
  size_t o0 = (size_t)blockIdx.x * Cc + t;
  if (outF) { outF[o0] = y0; outF[o0 + 256] = y1; }
  if (outB) { outB[o0] = f2bf(y0); outB[o0 + 256] = f2bf(y1); }
}

// ---------------- token-shift mix: 0.5*xm[t] + 0.5*xm[t-1] -> bf16 ----------------
__global__ void mix_shift(const float* __restrict__ xm, u16* __restrict__ mix) {
  int id = blockIdx.x * blockDim.x + threadIdx.x;
  if (id >= BT * Cc) return;
  int bt = id / Cc;
  int t = bt % Tt;
  float cur = xm[id];
  float prev = (t > 0) ? xm[id - Cc] : 0.f;
  mix[id] = f2bf(0.5f * cur + 0.5f * prev);
}

// ---------------- weight convert / convert+transpose to bf16 ----------------
__global__ void convert_f2b(const float* __restrict__ in, u16* __restrict__ out, int n) {
  int id = blockIdx.x * blockDim.x + threadIdx.x;
  if (id < n) out[id] = f2bf(in[id]);
}

// in: [K,N] row-major f32 -> out: [N,K] row-major bf16
__global__ void convert_transpose(const float* __restrict__ in, u16* __restrict__ out,
                                  int K, int N) {
  int id = blockIdx.x * blockDim.x + threadIdx.x;
  if (id >= K * N) return;
  int n = id / K, k = id % K;
  out[id] = f2bf(in[(size_t)k * N + n]);
}

// ---------------- WKV chunked linear-recurrence scan ----------------
// h_t = d*h_{t-1} + k_t*v_t ; y_t = sigmoid(q_t)*h_t ; d constant per channel.
__global__ void wkv_pass1(const float* __restrict__ k, const float* __restrict__ v,
                          const float* __restrict__ td, float* __restrict__ csum) {
  int b = blockIdx.x / NCH, ch = blockIdx.x % NCH;
  int c = threadIdx.x;                       // 512 threads, coalesced over C
  float d = expf(-expf(td[c]));
  float h = 0.f;
  size_t base = ((size_t)b * Tt + (size_t)ch * CL) * Cc + c;
  for (int t = 0; t < CL; ++t) {
    size_t o = base + (size_t)t * Cc;
    h = d * h + k[o] * v[o];
  }
  csum[((size_t)b * NCH + ch) * Cc + c] = h;
}

__global__ void wkv_pass2(const float* __restrict__ td, const float* __restrict__ csum,
                          float* __restrict__ hin, float* __restrict__ hout) {
  int id = blockIdx.x * blockDim.x + threadIdx.x;
  if (id >= Bn * Cc) return;
  int b = id / Cc, c = id % Cc;
  float d = expf(-expf(td[c]));
  float d2 = d * d, d4 = d2 * d2, d8 = d4 * d4, d16 = d8 * d8;   // d^CL
  float h = 0.f;
  for (int ch = 0; ch < NCH; ++ch) {
    size_t o = ((size_t)b * NCH + ch) * Cc + c;
    hin[o] = h;
    h = d16 * h + csum[o];
  }
  hout[id] = h;                               // final state for this layer
}

__global__ void wkv_pass3(const float* __restrict__ q, const float* __restrict__ k,
                          const float* __restrict__ v, const float* __restrict__ td,
                          const float* __restrict__ hin, u16* __restrict__ ybf) {
  int b = blockIdx.x / NCH, ch = blockIdx.x % NCH;
  int c = threadIdx.x;
  float d = expf(-expf(td[c]));
  float h = hin[((size_t)b * NCH + ch) * Cc + c];
  size_t base = ((size_t)b * Tt + (size_t)ch * CL) * Cc + c;
  for (int t = 0; t < CL; ++t) {
    size_t o = base + (size_t)t * Cc;
    h = d * h + k[o] * v[o];
    float sg = 1.f / (1.f + expf(-q[o]));
    ybf[o] = f2bf(sg * h);
  }
}

// ---------------- register-blocked, double-buffered WMMA GEMM ----------------
// out = act(A[M,K] @ Bt[N,K]^T + bias) (+ resid). A, Bt bf16 row-major.
// Wave computes a 64(M) x 32(N) macro-tile: 4x2 WMMA tiles, 8 accumulators.
// K loop is software-pipelined depth 2. __launch_bounds__(256,1) grants the
// ~180 VGPRs needed so acc/fragments stay in registers (no scratch spills);
// 8 waves/WGP + in-wave pipelining cover the load latency.
// act: 0=none, 1=gelu, 2=relu. ntout: nontemporal f32 stores (streaming logits).
#define TMB 4
#define TNB 2

struct Frag {
  union { v16bf v; v8bf h[2]; } a[TMB];
  union { v16bf v; v8bf h[2]; } b[TNB];
};

__device__ __forceinline__ void load_frags(Frag& f, const u16* const* arow,
                                           const u16* const* brow, int kc, int hi) {
#pragma unroll
  for (int j = 0; j < TNB; ++j) {
    // B 32x16 layout: elems 0..7 -> K = kc+8*hi.. ; elems 8..15 -> K = kc+16+8*hi..
    f.b[j].h[0] = *(const v8bf*)(brow[j] + kc + 8 * hi);
    f.b[j].h[1] = *(const v8bf*)(brow[j] + kc + 16 + 8 * hi);
  }
#pragma unroll
  for (int i = 0; i < TMB; ++i) {
    // A 16x32 layout: elems 0..7 -> K = kc+8*hi.. ; elems 8..15 -> K = kc+16+8*hi..
    f.a[i].h[0] = *(const v8bf*)(arow[i] + kc + 8 * hi);
    f.a[i].h[1] = *(const v8bf*)(arow[i] + kc + 16 + 8 * hi);
  }
}

__device__ __forceinline__ void do_wmmas(v8f acc[TMB][TNB], const Frag& f) {
#pragma unroll
  for (int i = 0; i < TMB; ++i)
#pragma unroll
    for (int j = 0; j < TNB; ++j)
      acc[i][j] = __builtin_amdgcn_wmma_f32_16x16x32_bf16(
          false, f.a[i].v, false, f.b[j].v, (short)0, acc[i][j], false, false);
}

__global__ void __launch_bounds__(256, 1)
gemm_wmma(const u16* __restrict__ A, const u16* __restrict__ Bt,
          float* __restrict__ outF, u16* __restrict__ outB,
          const float* __restrict__ bias, const float* __restrict__ resid,
          int M, int N, int K, int act, int ntout) {
  int wave = blockIdx.x * (blockDim.x >> 5) + (threadIdx.x >> 5);
  int lane = threadIdx.x & 31;
  int nt = N >> 5;                       // macro-tiles along N
  int tot = (M >> 6) * nt;
  if (wave >= tot) return;
  int wm = wave / nt, wn = wave % nt;
  int hi = lane >> 4, lo = lane & 15;

  const u16* arow[TMB];
  const u16* brow[TNB];
#pragma unroll
  for (int i = 0; i < TMB; ++i) arow[i] = A + (size_t)(wm * 64 + i * 16 + lo) * K;
#pragma unroll
  for (int j = 0; j < TNB; ++j) brow[j] = Bt + (size_t)(wn * 32 + j * 16 + lo) * K;

  v8f acc[TMB][TNB];
#pragma unroll
  for (int i = 0; i < TMB; ++i)
#pragma unroll
    for (int j = 0; j < TNB; ++j) acc[i][j] = (v8f){};

  Frag f0, f1;
  load_frags(f0, arow, brow, 0, hi);
  int kc = 0;
  for (; kc + 64 <= K; kc += 64) {
    load_frags(f1, arow, brow, kc + 32, hi);   // in flight during f0 math
    do_wmmas(acc, f0);
    if (kc + 64 < K) load_frags(f0, arow, brow, kc + 64, hi);
    do_wmmas(acc, f1);
  }
  if (kc < K) do_wmmas(acc, f0);               // K % 64 == 32 tail

#pragma unroll
  for (int i = 0; i < TMB; ++i)
#pragma unroll
    for (int j = 0; j < TNB; ++j) {
      int col = wn * 32 + j * 16 + lo;
      float bv = bias ? bias[col] : 0.f;
#pragma unroll
      for (int r = 0; r < 8; ++r) {
        int row = wm * 64 + i * 16 + hi * 8 + r;   // C/D layout: vgpr r + lane half
        float v = acc[i][j][r] + bv;
        if (act == 1)      v = 0.5f * v * (1.f + erff(v * 0.70710678118f));
        else if (act == 2) v = fmaxf(v, 0.f);
        size_t o = (size_t)row * N + col;
        if (resid) v += resid[o];
        if (outF) {
          if (ntout) __builtin_nontemporal_store(v, outF + o);
          else       outF[o] = v;
        }
        if (outB) outB[o] = f2bf(v);
      }
    }
}

// ---------------- host orchestration ----------------
extern "C" void kernel_launch(void* const* d_in, const int* in_sizes, int n_in,
                              void* d_out_v, int out_size, void* d_ws, size_t ws_size,
                              hipStream_t stream) {
  const int*   ids     = (const int*)d_in[0];
  const float* embed   = (const float*)d_in[1];
  const float* ln1_g   = (const float*)d_in[2];
  const float* ln1_b   = (const float*)d_in[3];
  const float* tm_g    = (const float*)d_in[4];
  const float* tm_b    = (const float*)d_in[5];
  const float* q_a     = (const float*)d_in[6];
  const float* q_b     = (const float*)d_in[7];
  const float* k_a     = (const float*)d_in[8];
  const float* k_b     = (const float*)d_in[9];
  const float* v_a     = (const float*)d_in[10];
  const float* v_b     = (const float*)d_in[11];
  const float* out_w   = (const float*)d_in[12];
  const float* out_b   = (const float*)d_in[13];
  const float* tdecay  = (const float*)d_in[14];
  const float* ln2_g   = (const float*)d_in[15];
  const float* ln2_b   = (const float*)d_in[16];
  const float* ad_down = (const float*)d_in[17];
  const float* ad_up   = (const float*)d_in[18];
  const float* ffn_w1  = (const float*)d_in[19];
  const float* ffn_b1  = (const float*)d_in[20];
  const float* ffn_w2  = (const float*)d_in[21];
  const float* ffn_b2  = (const float*)d_in[22];
  const float* lnf_g   = (const float*)d_in[23];
  const float* lnf_b   = (const float*)d_in[24];

  float* d_out = (float*)d_out_v;
  const size_t logitsN = (size_t)BT * Vv;

  // workspace carve-up (~125 MB)
  char* ws = (char*)d_ws;
  size_t off = 0;
  auto alloc = [&](size_t bytes) { size_t o = off; off += (bytes + 255) & ~(size_t)255; return o; };
  float* X    = (float*)(ws + alloc((size_t)BT * Cc * 4));  // residual stream
  float* XMV  = (float*)(ws + alloc((size_t)BT * Cc * 4));  // xm, later reused as V
  float* Qf   = (float*)(ws + alloc((size_t)BT * Cc * 4));
  float* Kf   = (float*)(ws + alloc((size_t)BT * Cc * 4));
  u16*   BUF1 = (u16*)(ws + alloc((size_t)BT * Cc * 2));    // mix / x2 / adapter-in (bf16)
  u16*   BUF2 = (u16*)(ws + alloc((size_t)BT * Cc * 2));    // y / final-ln (bf16)
  u16*   T1B  = (u16*)(ws + alloc((size_t)BT * Rr * 2));    // low-rank mids (bf16)
  u16*   H1B  = (u16*)(ws + alloc((size_t)BT * FFNc * 2));  // FFN hidden (bf16)
  u16*   WT   = (u16*)(ws + alloc((size_t)Vv * Cc * 2));    // weight bf16 scratch (8 MB)
  float* CSUM = (float*)(ws + alloc((size_t)Bn * NCH * Cc * 4)); // WKV chunk sums
  float* HIN  = (float*)(ws + alloc((size_t)Bn * NCH * Cc * 4)); // WKV chunk-entry states
  u16*   WTA  = WT;
  u16*   WTB  = WT + 2 * 1024 * 1024;

  auto g1 = [](size_t n) { return dim3((unsigned)((n + 255) / 256)); };
  auto gemm = [&](const u16* A, const u16* Btm, float* oF, u16* oB,
                  const float* bias, const float* resid, int M, int N, int K,
                  int act, int ntout) {
    int tiles = (M / 64) * (N / 32);
    gemm_wmma<<<dim3((unsigned)((tiles + 7) / 8)), dim3(256), 0, stream>>>(
        A, Btm, oF, oB, bias, resid, M, N, K, act, ntout);
  };
  auto ctr = [&](const float* in, u16* out, int K, int N) {
    convert_transpose<<<g1((size_t)K * N), dim3(256), 0, stream>>>(in, out, K, N);
  };

  gather_embed<<<g1((size_t)BT * Cc), dim3(256), 0, stream>>>(ids, embed, X);

  for (int i = 0; i < Ll; ++i) {
    // --- time mixing ---
    ln2x_kernel<<<dim3(BT), dim3(256), 0, stream>>>(
        X, ln1_g + i * Cc, ln1_b + i * Cc, tm_g + i * Cc, tm_b + i * Cc, XMV);
    mix_shift<<<g1((size_t)BT * Cc), dim3(256), 0, stream>>>(XMV, BUF1);

    const float* pa[3] = { q_a + (size_t)i * Cc * Rr, k_a + (size_t)i * Cc * Rr,
                           v_a + (size_t)i * Cc * Rr };
    const float* pb[3] = { q_b + (size_t)i * Rr * Cc, k_b + (size_t)i * Rr * Cc,
                           v_b + (size_t)i * Rr * Cc };
    float* po[3] = { Qf, Kf, XMV };  // V overwrites xm (dead after mix)
    for (int p = 0; p < 3; ++p) {
      ctr(pa[p], WTA, Cc, Rr);                                            // [C,R] -> [R,C]
      gemm(BUF1, WTA, nullptr, T1B, nullptr, nullptr, BT, Rr, Cc, 0, 0);  // mid (bf16)
      ctr(pb[p], WTB, Rr, Cc);                                            // [R,C] -> [C,R]
      gemm(T1B, WTB, po[p], nullptr, nullptr, nullptr, BT, Cc, Rr, 0, 0);
    }

    // --- WKV: chunked parallel linear scan ---
    const float* td = tdecay + i * Cc;
    wkv_pass1<<<dim3(Bn * NCH), dim3(Cc), 0, stream>>>(Kf, XMV, td, CSUM);
    wkv_pass2<<<g1((size_t)Bn * Cc), dim3(256), 0, stream>>>(
        td, CSUM, HIN, d_out + logitsN + (size_t)i * Bn * Cc);
    wkv_pass3<<<dim3(Bn * NCH), dim3(Cc), 0, stream>>>(Qf, Kf, XMV, td, HIN, BUF2);

    // x += y @ out_w + out_b
    ctr(out_w + (size_t)i * Cc * Cc, WTA, Cc, Cc);
    gemm(BUF2, WTA, X, nullptr, out_b + i * Cc, X, BT, Cc, Cc, 0, 0);

    // --- shared FFN ---
    ln_kernel<<<dim3(BT), dim3(256), 0, stream>>>(X, ln2_g + i * Cc, ln2_b + i * Cc,
                                                  nullptr, BUF1);
    int g = i / GRP;
    ctr(ffn_w1 + (size_t)g * Cc * FFNc, WTA, Cc, FFNc);
    gemm(BUF1, WTA, nullptr, H1B, ffn_b1 + g * FFNc, nullptr, BT, FFNc, Cc, 1, 0);
    ctr(ffn_w2 + (size_t)g * FFNc * Cc, WTB, FFNc, Cc);
    gemm(H1B, WTB, X, nullptr, ffn_b2 + g * Cc, X, BT, Cc, FFNc, 0, 0);

    // --- adapter: x += relu(x @ ad_down) @ ad_up ---
    convert_f2b<<<g1((size_t)BT * Cc), dim3(256), 0, stream>>>(X, BUF1, BT * Cc);
    ctr(ad_down + (size_t)i * Cc * ARr, WTA, Cc, ARr);
    gemm(BUF1, WTA, nullptr, T1B, nullptr, nullptr, BT, ARr, Cc, 2, 0);
    ctr(ad_up + (size_t)i * ARr * Cc, WTB, ARr, Cc);
    gemm(T1B, WTB, X, nullptr, nullptr, X, BT, Cc, ARr, 0, 0);
  }

  // --- final LN + tied head: logits = ln(x) @ embed^T (nontemporal stores) ---
  ln_kernel<<<dim3(BT), dim3(256), 0, stream>>>(X, lnf_g, lnf_b, nullptr, BUF2);
  convert_f2b<<<g1((size_t)Vv * Cc), dim3(256), 0, stream>>>(embed, WT, Vv * Cc);
  gemm(BUF2, WT, d_out, nullptr, nullptr, nullptr, BT, Vv, Cc, 0, 1);
}
